// AMEGNN_48034914239012
// MI455X (gfx1250) — matrix operand order repaired and checked by reading, not compiled, not run on hardware
//
#include <hip/hip_runtime.h>
#include <hip/hip_fp16.h>

typedef __attribute__((ext_vector_type(16))) _Float16 v16h;
typedef __attribute__((ext_vector_type(8)))  float    v8f;

#define NCH 4
#define NA 16
#define DOUT 64
#define FDIM 273      // A*A + 2*C + 9
#define KPAD 288      // padded to multiple of 32
#define OUTSTRIDE 76  // 64 radials + 12 coord_agg

// ---------------------------------------------------------------------------
// Per-node local frames; also zero the per-node edge counter.
// ---------------------------------------------------------------------------
__global__ void lf_cnt_kernel(const float* __restrict__ coord,
                              float* __restrict__ lf,
                              float* __restrict__ cnt, int n_nodes) {
    int n = blockIdx.x * blockDim.x + threadIdx.x;
    if (n >= n_nodes) return;
    const float* p = coord + (size_t)n * (NCH * 3);
    float ax = p[0], ay = p[1], az = p[2];
    float bx = p[3] - ax, by = p[4] - ay, bz = p[5] - az;
    float tn = sqrtf(bx * bx + by * by + bz * bz) + 1e-8f;
    bx /= tn; by /= tn; bz /= tn;
    float tx = p[6] - ax, ty = p[7] - ay, tz = p[8] - az;
    float d = tx * bx + ty * by + tz * bz;
    float yx = tx - d * bx, yy = ty - d * by, yz = tz - d * bz;
    float yn = sqrtf(yx * yx + yy * yy + yz * yz) + 1e-8f;
    yx /= yn; yy /= yn; yz /= yn;
    float zx = by * yz - bz * yy;
    float zy = bz * yx - bx * yz;
    float zz = bx * yy - by * yx;
    float* o = lf + (size_t)n * 9;
    o[0] = bx; o[1] = yx; o[2] = zx;
    o[3] = by; o[4] = yy; o[5] = zy;
    o[6] = bz; o[7] = yz; o[8] = zz;
    cnt[n] = 0.0f;
}

__global__ void zero_out_kernel(float* __restrict__ out, int total) {
    int t = blockIdx.x * blockDim.x + threadIdx.x;
    if (t < total) out[t] = 0.0f;
}

// W (64 x 273 f32) -> Wh (64 x 288 f16), zero padded.
__global__ void wconv_kernel(const float* __restrict__ W, _Float16* __restrict__ Wh) {
    int t = blockIdx.x * blockDim.x + threadIdx.x;
    if (t >= DOUT * KPAD) return;
    int n = t / KPAD, k = t % KPAD;
    Wh[t] = (k < FDIM) ? (_Float16)W[n * FDIM + k] : (_Float16)0.0f;
}

// ---------------------------------------------------------------------------
// Main edge kernel: 128 threads = 4 waves, 16 edges per wave.
// ---------------------------------------------------------------------------
__launch_bounds__(128)
__global__ void edge_kernel(const float* __restrict__ coord,
                            const float* __restrict__ attr,
                            const float* __restrict__ cw,
                            const _Float16* __restrict__ Wh,
                            const float* __restrict__ bvec,
                            const int* __restrict__ row,
                            const int* __restrict__ col,
                            const float* __restrict__ lf,
                            float* __restrict__ out,
                            float* __restrict__ cnt,
                            int n_edges) {
    __shared__ _Float16 lA[4][16 * KPAD];  // per-wave 16x288 f16 A-tiles
    __shared__ float    lRn[4][16];        // per-edge 1/rnorm
    __shared__ int      lRow[4][16];       // per-edge destination node (-1 = invalid)

    const int lane = threadIdx.x & 31;
    const int wave = threadIdx.x >> 5;

    // ================= build phase: 2 lanes per edge =================
    const int m    = lane >> 1;   // edge slot 0..15
    const int half = lane & 1;    // which 8 rows of the 16x16 radial block
    const int e    = blockIdx.x * 64 + wave * 16 + m;
    const bool valid = (e < n_edges);

    int ni = 0, nj = 0;
    if (valid) { ni = row[e]; nj = col[e]; }

    float ci[NCH][3], cj[NCH][3], wwi[NCH], wwj[NCH];
#pragma unroll
    for (int c = 0; c < NCH; ++c) {
#pragma unroll
        for (int d = 0; d < 3; ++d) {
            ci[c][d] = valid ? coord[(ni * NCH + c) * 3 + d] : 0.0f;
            cj[c][d] = valid ? coord[(nj * NCH + c) * 3 + d] : 0.0f;
        }
        wwi[c] = valid ? cw[ni * NCH + c] : 0.0f;
        wwj[c] = valid ? cw[nj * NCH + c] : 0.0f;
    }

    // coord_msg[c1][c2] = ||ci[c1]-cj[c2]|| * wi[c1]*wj[c2]
    float cm[NCH][NCH];
#pragma unroll
    for (int c1 = 0; c1 < NCH; ++c1)
#pragma unroll
        for (int c2 = 0; c2 < NCH; ++c2) {
            float dx = ci[c1][0] - cj[c2][0];
            float dy = ci[c1][1] - cj[c2][1];
            float dz = ci[c1][2] - cj[c2][2];
            cm[c1][c2] = sqrtf(dx * dx + dy * dy + dz * dz) * wwi[c1] * wwj[c2];
        }

    // tmp[a][c2] = sum_c1 attr_i[c1][a0+a] * cm[c1][c2]
    const int a0 = half * 8;
    float tmp[8][NCH];
#pragma unroll
    for (int a = 0; a < 8; ++a)
#pragma unroll
        for (int c2 = 0; c2 < NCH; ++c2) tmp[a][c2] = 0.0f;
#pragma unroll
    for (int c1 = 0; c1 < NCH; ++c1) {
        float aiv[8];
#pragma unroll
        for (int a = 0; a < 8; ++a)
            aiv[a] = valid ? attr[(ni * NCH + c1) * NA + a0 + a] : 0.0f;
#pragma unroll
        for (int a = 0; a < 8; ++a)
#pragma unroll
            for (int c2 = 0; c2 < NCH; ++c2) tmp[a][c2] += aiv[a] * cm[c1][c2];
    }

    // rad[a][b] = sum_c2 tmp[a][c2] * attr_j[c2][b]
    float rad[8][NA];
#pragma unroll
    for (int a = 0; a < 8; ++a)
#pragma unroll
        for (int bb = 0; bb < NA; ++bb) rad[a][bb] = 0.0f;
#pragma unroll
    for (int c2 = 0; c2 < NCH; ++c2) {
        float ajv[NA];
#pragma unroll
        for (int bb = 0; bb < NA; ++bb)
            ajv[bb] = valid ? attr[(nj * NCH + c2) * NA + bb] : 0.0f;
#pragma unroll
        for (int a = 0; a < 8; ++a) {
            float t = tmp[a][c2];
#pragma unroll
            for (int bb = 0; bb < NA; ++bb) rad[a][bb] += t * ajv[bb];
        }
    }

    // write my 8x16 radial slab to LDS (f16) and accumulate sum of squares
    float ssq = 0.0f;
    _Float16* rowp = &lA[wave][m * KPAD];
#pragma unroll
    for (int a = 0; a < 8; ++a)
#pragma unroll
        for (int bb = 0; bb < NA; ++bb) {
            float v = rad[a][bb];
            ssq += v * v;
            rowp[(a0 + a) * NA + bb] = (_Float16)v;
        }

    float elf[9];
#pragma unroll
    for (int k = 0; k < 9; ++k) elf[k] = 0.0f;
    if (half == 0) {
#pragma unroll
        for (int k = 0; k < 9; ++k) {
            float v = valid ? (lf[ni * 9 + k] + lf[nj * 9 + k]) : 0.0f;
            elf[k] = v;
            ssq += v * v;
        }
    }
    float tot = ssq + __shfl_xor(ssq, 1, 32);

    if (half == 0) {
        lRn[wave][m]  = 1.0f / (sqrtf(tot) + 1.0f);
        lRow[wave][m] = valid ? ni : -1;
#pragma unroll
        for (int k = 0; k < 8; ++k) rowp[256 + k] = (_Float16)0.0f;   // zeros(2C)
#pragma unroll
        for (int k = 0; k < 9; ++k) rowp[264 + k] = (_Float16)elf[k]; // edge_lf
#pragma unroll
        for (int k = FDIM; k < KPAD; ++k) rowp[k] = (_Float16)0.0f;   // pad

        if (valid) {
            // pooled_col and coord_diff accumulation
            float csum = 0.0f, mj[NCH];
#pragma unroll
            for (int c = 0; c < NCH; ++c) {
                mj[c] = (wwj[c] != 0.0f) ? 1.0f : 0.0f;
                csum += mj[c];
            }
            float px = 0.f, py = 0.f, pz = 0.f;
#pragma unroll
            for (int c = 0; c < NCH; ++c) {
                px += cj[c][0] * mj[c];
                py += cj[c][1] * mj[c];
                pz += cj[c][2] * mj[c];
            }
            float inv_cs = 1.0f / csum;
            px *= inv_cs; py *= inv_cs; pz *= inv_cs;
            float* op = out + (size_t)ni * OUTSTRIDE + 64;
#pragma unroll
            for (int c = 0; c < NCH; ++c) {
                atomicAdd(op + c * 3 + 0, ci[c][0] - px);
                atomicAdd(op + c * 3 + 1, ci[c][1] - py);
                atomicAdd(op + c * 3 + 2, ci[c][2] - pz);
            }
            atomicAdd(cnt + ni, 1.0f);
        }
    }

    __syncthreads();

    // ================= WMMA phase: 16x288 @ 288x64 -> 16x64 =================
    // A operand, contiguous k order used here (not ISA table order): lane l holds row M=l&15; kh=l>>4:
    //   halves 0..15 -> K = 32*kb + 16*kh + i
    // B operand, same contiguous k order as A: lane l holds col N=l&15; kh=l>>4:
    //   halves 0..15 -> K = 32*kb + 16*kh + i
    const int aM  = lane & 15;
    const int akh = lane >> 4;

    v8f acc[4];
#pragma unroll
    for (int nt = 0; nt < 4; ++nt)
#pragma unroll
        for (int r = 0; r < 8; ++r) acc[nt][r] = 0.0f;

#pragma unroll
    for (int kb = 0; kb < KPAD / 32; ++kb) {
        v16h af;
        const _Float16* p0 = &lA[wave][aM * KPAD + kb * 32 + akh * 16];
#pragma unroll
        for (int i = 0; i < 8; ++i) {
            af[i]     = p0[i];
            af[8 + i] = p0[8 + i];
        }
#pragma unroll
        for (int nt = 0; nt < 4; ++nt) {
            v16h bf;
            const _Float16* q = &Wh[(nt * 16 + aM) * KPAD + kb * 32 + akh * 16];
#pragma unroll
            for (int i = 0; i < 16; ++i) bf[i] = q[i];
            acc[nt] = __builtin_amdgcn_wmma_f32_16x16x32_f16(
                false, af, false, bf, (short)0, acc[nt], false, false);
        }
    }

    // C/D layout: VGPR r, lane l -> M = 8*(l>>4) + r, N = l&15
#pragma unroll
    for (int nt = 0; nt < 4; ++nt) {
        float bv = bvec[nt * 16 + aM];
#pragma unroll
        for (int r = 0; r < 8; ++r) {
            int mm   = akh * 8 + r;
            int ridx = lRow[wave][mm];
            if (ridx >= 0) {
                float v = (acc[nt][r] + bv) * lRn[wave][mm];
                atomicAdd(out + (size_t)ridx * OUTSTRIDE + nt * 16 + aM, v);
            }
        }
    }
}

// Scale coord_agg slice by 1/max(cnt,1).
__global__ void finalize_kernel(float* __restrict__ out,
                                const float* __restrict__ cnt, int n_nodes) {
    int n = blockIdx.x * blockDim.x + threadIdx.x;
    if (n >= n_nodes) return;
    float s = 1.0f / fmaxf(cnt[n], 1.0f);
    float* p = out + (size_t)n * OUTSTRIDE + 64;
#pragma unroll
    for (int k = 0; k < 12; ++k) p[k] *= s;
}

// ---------------------------------------------------------------------------
extern "C" void kernel_launch(void* const* d_in, const int* in_sizes, int n_in,
                              void* d_out, int out_size, void* d_ws, size_t ws_size,
                              hipStream_t stream) {
    const float* coord = (const float*)d_in[0];
    const float* attr  = (const float*)d_in[1];
    const float* cw    = (const float*)d_in[2];
    const float* W     = (const float*)d_in[3];
    const float* bvec  = (const float*)d_in[4];
    const int*   row   = (const int*)d_in[5];
    const int*   col   = (const int*)d_in[6];
    float* out = (float*)d_out;

    const int n_nodes = in_sizes[2] / NCH;  // channel_weights is N*C
    const int n_edges = in_sizes[5];

    // workspace layout: lf (N*9 f32) | cnt (N f32) | Wh (64*288 f16)
    float* lf  = (float*)d_ws;
    float* cnt = lf + (size_t)n_nodes * 9;
    _Float16* Wh = (_Float16*)(cnt + n_nodes);  // 16B-aligned for N*10 % 4 == 0

    const int tot_out = n_nodes * OUTSTRIDE;
    zero_out_kernel<<<(tot_out + 255) / 256, 256, 0, stream>>>(out, tot_out);
    lf_cnt_kernel<<<(n_nodes + 255) / 256, 256, 0, stream>>>(coord, lf, cnt, n_nodes);
    wconv_kernel<<<(DOUT * KPAD + 255) / 256, 256, 0, stream>>>(W, Wh);

    const int nblocks = (n_edges + 63) / 64;  // 64 edges per 128-thread block
    edge_kernel<<<nblocks, 128, 0, stream>>>(coord, attr, cw, Wh, bvec, row, col,
                                             lf, out, cnt, n_edges);
    finalize_kernel<<<(n_nodes + 255) / 256, 256, 0, stream>>>(out, cnt, n_nodes);
}
